// MambaBlock_3547642986681
// MI455X (gfx1250) — matrix operand order repaired and checked by reading, not compiled, not run on hardware
//
#include <hip/hip_runtime.h>

// ---------------- problem constants (from reference) ----------------
#define BATCH   4
#define SEQ     2048
#define DMODEL  1024
#define DINNER  2048          // 2*DMODEL
#define DSTATE  16
#define DCONV   4
#define DTRANK  64            // DMODEL/16
#define ROWS    (BATCH*SEQ)   // 8192
#define XPROJ_LD 128          // dt_rank + 2*d_state = 96, padded to 128

typedef _Float16 half_t;
typedef __attribute__((ext_vector_type(8)))  _Float16 v8h;
typedef __attribute__((ext_vector_type(16))) _Float16 v16h;
typedef __attribute__((ext_vector_type(8)))  float    v8f;

__device__ __forceinline__ float softplusf(float x) {
    return x > 20.f ? x : log1pf(__expf(x));
}
__device__ __forceinline__ float siluf(float x) {
    return x / (1.f + __expf(-x));
}

// ---------------- weight preparation kernels (once per launch) ----------------
// in: K x N row-major f32;  out: N x K row-major f16 (transposed weight)
__global__ __launch_bounds__(256) void transpose_cast_kernel(
    const float* __restrict__ in, half_t* __restrict__ out, int K, int N) {
    int i = blockIdx.x * 256 + threadIdx.x;
    if (i >= K * N) return;
    int n = i / K, k = i % K;                 // output-linear (coalesced writes)
    out[i] = (half_t)in[(size_t)k * N + n];
}

// W_x is (DINNER x 96); produce transposed+padded (128 x DINNER) f16
__global__ __launch_bounds__(256) void transpose_pad_wx_kernel(
    const float* __restrict__ wx, half_t* __restrict__ out) {
    int i = blockIdx.x * 256 + threadIdx.x;
    if (i >= XPROJ_LD * DINNER) return;
    int n = i / DINNER, k = i % DINNER;
    out[i] = (n < (DTRANK + 2 * DSTATE)) ? (half_t)wx[(size_t)k * (DTRANK + 2 * DSTATE) + n]
                                         : (half_t)0.f;
}

// extract dt-rank columns of dbl (ld=128) to a dense f16 (ROWS x 64) matrix
__global__ __launch_bounds__(256) void extract_dt_kernel(
    const float* __restrict__ dbl, half_t* __restrict__ out) {
    int i = blockIdx.x * 256 + threadIdx.x;
    if (i >= ROWS * DTRANK) return;
    int r = i / DTRANK, c = i % DTRANK;
    out[i] = (half_t)dbl[(size_t)r * XPROJ_LD + c];
}

// ---------------- LayerNorm -> f16 ----------------
__global__ __launch_bounds__(256) void layernorm_kernel(
    const float* __restrict__ x, const float* __restrict__ gamma,
    const float* __restrict__ beta, half_t* __restrict__ out) {
    __shared__ float red[2][8];
    const int row = blockIdx.x;
    const float* xr = x + (size_t)row * DMODEL;
    float s = 0.f, s2 = 0.f;
    for (int c = threadIdx.x; c < DMODEL; c += 256) {
        float v = xr[c]; s += v; s2 += v * v;
    }
    #pragma unroll
    for (int off = 16; off > 0; off >>= 1) {  // wave32 reduction
        s  += __shfl_down(s,  off, 32);
        s2 += __shfl_down(s2, off, 32);
    }
    const int lane = threadIdx.x & 31, wave = threadIdx.x >> 5;
    if (lane == 0) { red[0][wave] = s; red[1][wave] = s2; }
    __syncthreads();
    if (wave == 0) {
        s  = (lane < 8) ? red[0][lane] : 0.f;
        s2 = (lane < 8) ? red[1][lane] : 0.f;
        #pragma unroll
        for (int off = 4; off > 0; off >>= 1) {
            s  += __shfl_down(s,  off, 32);
            s2 += __shfl_down(s2, off, 32);
        }
        if (lane == 0) { red[0][0] = s; red[1][0] = s2; }
    }
    __syncthreads();
    const float mu  = red[0][0] * (1.f / DMODEL);
    const float var = red[1][0] * (1.f / DMODEL) - mu * mu;
    const float rs  = rsqrtf(var + 1e-5f);
    for (int c = threadIdx.x; c < DMODEL; c += 256)
        out[(size_t)row * DMODEL + c] = (half_t)((xr[c] - mu) * rs * gamma[c] + beta[c]);
}

// ---------------- WMMA f16 GEMM: C(M,N) = A(M,K) * Bt(N,K)^T, f32 accum ----------
// 128x64 block tile, BK=32, 8 wave32 waves; each wave owns a 32x32 macro-tile
// (2x2 of 16x16) -> 4 v_wmma per K-step with A/B fragment reuse.
// BOTH tiles staged with global_load_async_to_lds_b128 (ASYNCcnt path, no LDS
// stores at all), double-buffered and software-pipelined: prefetch tile i+1,
// s_wait_asynccnt 3 (3 async instructions per wave per tile) drains tile i.
#define BM 128
#define BN 64
#define BK 32
#define LDA_S 40   // halves per A-tile row in LDS (32 + 8 pad, 80B: 16B-aligned rows)
#define LDBT  40   // halves per Bt row (K-dim) in LDS

enum { EPI_NONE = 0, EPI_SOFTPLUS_BIAS = 1, EPI_RESID = 2 };

template <int EPI>
__global__ __launch_bounds__(256) void wmma_gemm_kernel(
    const half_t* __restrict__ A, const half_t* __restrict__ Bt_g,
    float* __restrict__ C, int M, int N, int K, int lda, int ldb, int ldc,
    const float* __restrict__ bias, const float* __restrict__ resid) {
    __shared__ __align__(16) half_t As[2][BM * LDA_S];  // 2 x 10240 B
    __shared__ __align__(16) half_t Bt[2][BN * LDBT];   // 2 x  5120 B

    const int tid  = threadIdx.x;
    const int lane = tid & 31;
    const int wave = tid >> 5;
    const int bm   = blockIdx.y * BM;
    const int bn   = blockIdx.x * BN;
    const int wr   = (wave & 3) * 32;   // wave's 32-row band within the block tile
    const int wc   = (wave >> 2) * 32;  // wave's 32-col band
    const int g    = lane >> 4;         // half-wave group (ISA 7.12.2 16-bit layouts)
    const int mloc = lane & 15;

    const unsigned as_base = (unsigned)(uintptr_t)(&As[0][0]);
    const unsigned bt_base = (unsigned)(uintptr_t)(&Bt[0][0]);
    const unsigned as_sz   = (unsigned)(BM * LDA_S * 2);  // bytes per buffer
    const unsigned bt_sz   = (unsigned)(BN * LDBT * 2);

    // stage one K-tile (3 async b128 instructions per thread/wave)
    auto stage = [&](int k0, int buf) {
        #pragma unroll
        for (int c = 0; c < 2; ++c) {
            const int chunk = tid + c * 256;        // 512 chunks of 8 halves
            const int r     = chunk >> 2;           // 0..127
            const int co    = (chunk & 3) * 8;      // 0,8,16,24
            const unsigned la =
                as_base + (unsigned)buf * as_sz + (unsigned)(r * LDA_S + co) * 2u;
            const unsigned long long ga =
                (unsigned long long)(A + (size_t)(bm + r) * lda + (k0 + co));
            asm volatile("global_load_async_to_lds_b128 %0, %1, off"
                         :: "v"(la), "v"(ga) : "memory");
        }
        {
            const int n  = tid >> 2;                // 0..63
            const int co = (tid & 3) * 8;           // 0,8,16,24
            const unsigned la =
                bt_base + (unsigned)buf * bt_sz + (unsigned)(n * LDBT + co) * 2u;
            const unsigned long long ga =
                (unsigned long long)(Bt_g + (size_t)(bn + n) * ldb + (k0 + co));
            asm volatile("global_load_async_to_lds_b128 %0, %1, off"
                         :: "v"(la), "v"(ga) : "memory");
        }
    };

    v8f acc[4] = {v8f{}, v8f{}, v8f{}, v8f{}};  // [rr*2+cc]

    const int nk = K / BK;
    stage(0, 0);  // prologue prefetch

    for (int i = 0; i < nk; ++i) {
        if (i + 1 < nk) {
            stage((i + 1) * BK, (i + 1) & 1);
            asm volatile("s_wait_asynccnt 0x3" ::: "memory");  // tile i landed
        } else {
            asm volatile("s_wait_asynccnt 0x0" ::: "memory");
        }
        __syncthreads();  // tile i visible to all waves

        const half_t* Asb = &As[i & 1][0];
        const half_t* Btb = &Bt[i & 1][0];

        // --- build fragments (two ds_load_b128 each) ---
        v16h af[2], bf[2];
        #pragma unroll
        for (int t = 0; t < 2; ++t) {
            const int row = wr + t * 16 + mloc;           // A: M row
            v8h alo = *(const v8h*)&Asb[row * LDA_S + g * 8];       // k = g*8 + 0..7
            v8h ahi = *(const v8h*)&Asb[row * LDA_S + 16 + g * 8];  // k = 16+g*8+0..7
            af[t] = __builtin_shufflevector(alo, ahi, 0, 1, 2, 3, 4, 5, 6, 7,
                                            8, 9, 10, 11, 12, 13, 14, 15);
            const int col = wc + t * 16 + mloc;           // B: N col
            v8h blo = *(const v8h*)&Btb[col * LDBT + g * 8];        // k = g*8 + 0..7
            v8h bhi = *(const v8h*)&Btb[col * LDBT + 16 + g * 8];   // k = 16+g*8+0..7
            bf[t] = __builtin_shufflevector(blo, bhi, 0, 1, 2, 3, 4, 5, 6, 7,
                                            8, 9, 10, 11, 12, 13, 14, 15);
        }
        #pragma unroll
        for (int rr = 0; rr < 2; ++rr)
            #pragma unroll
            for (int cc = 0; cc < 2; ++cc)
                acc[rr * 2 + cc] = __builtin_amdgcn_wmma_f32_16x16x32_f16(
                    false, af[rr], false, bf[cc], (short)0, acc[rr * 2 + cc],
                    false, false);
        __syncthreads();  // done reading buf before it is overwritten again
    }

    // --- epilogue: D VGPR r holds row (r + 8*g), col = lane&15 within subtile ---
    #pragma unroll
    for (int rr = 0; rr < 2; ++rr) {
        #pragma unroll
        for (int cc = 0; cc < 2; ++cc) {
            #pragma unroll
            for (int r = 0; r < 8; ++r) {
                const int row = bm + wr + rr * 16 + r + (g << 3);
                const int col = bn + wc + cc * 16 + mloc;
                float v = acc[rr * 2 + cc][r];
                if (EPI == EPI_SOFTPLUS_BIAS) {
                    v = softplusf(v + bias[col]);
                } else if (EPI == EPI_RESID) {
                    v += resid[(size_t)row * ldc + col];
                }
                C[(size_t)row * ldc + col] = v;
            }
        }
    }
}

// ---------------- depthwise causal conv (k=4) + SiLU -> f16 ----------------
__global__ __launch_bounds__(256) void conv_silu_kernel(
    const float* __restrict__ xz, const float* __restrict__ cw,
    const float* __restrict__ cb, half_t* __restrict__ xc16) {
    const size_t idx = (size_t)blockIdx.x * 256 + threadIdx.x;
    if (idx >= (size_t)ROWS * DINNER) return;
    const int    d    = (int)(idx % DINNER);
    const size_t row  = idx / DINNER;
    const int    l    = (int)(row % SEQ);
    const size_t brow = row - l;  // first row of this batch element
    float acc = cb[d];
    #pragma unroll
    for (int k = 0; k < DCONV; ++k) {
        const int ls = l + k - (DCONV - 1);
        if (ls >= 0)
            acc += xz[(brow + ls) * (size_t)(2 * DINNER) + d] * cw[d * DCONV + k];
    }
    xc16[idx] = (half_t)siluf(acc);
}

// ---------------- selective scan (sequential over L) + gating -> f16 ----------------
// B/C for each timestep are loaded one-value-per-lane and broadcast with wave32
// shuffles (ds_bpermute) -- no barriers in the 2048-step loop.
__global__ __launch_bounds__(256) void scan_kernel(
    const float* __restrict__ dt, const half_t* __restrict__ xc16,
    const float* __restrict__ dbl, const float* __restrict__ xz,
    const float* __restrict__ A_log, const float* __restrict__ D_skip,
    half_t* __restrict__ y16) {
    const int b    = blockIdx.x / (DINNER / 256);
    const int d    = (blockIdx.x % (DINNER / 256)) * 256 + threadIdx.x;
    const int lane = threadIdx.x & 31;

    float Areg[DSTATE], h[DSTATE];
    #pragma unroll
    for (int n = 0; n < DSTATE; ++n) {
        Areg[n] = -__expf(A_log[d * DSTATE + n]);
        h[n]    = 0.f;
    }
    const float Dd = D_skip[d];

    for (int l = 0; l < SEQ; ++l) {
        const size_t row = (size_t)b * SEQ + l;
        // lanes 0..15 hold B[n], lanes 16..31 hold C[n]
        const float bc  = dbl[row * XPROJ_LD + DTRANK + lane];
        const float dtv = dt[row * DINNER + d];
        const float xv  = (float)xc16[row * DINNER + d];
        const float dx  = dtv * xv;
        float y = 0.f;
        #pragma unroll
        for (int n = 0; n < DSTATE; ++n) {
            const float Bn = __shfl(bc, n, 32);
            const float Cn = __shfl(bc, 16 + n, 32);
            const float e  = __expf(dtv * Areg[n]);
            h[n] = e * h[n] + dx * Bn;
            y += h[n] * Cn;
        }
        y += Dd * xv;
        const float zv = xz[row * (size_t)(2 * DINNER) + DINNER + d];
        y *= siluf(zv);
        y16[row * DINNER + d] = (half_t)y;
    }
}

// ---------------- launch ----------------
extern "C" void kernel_launch(void* const* d_in, const int* in_sizes, int n_in,
                              void* d_out, int out_size, void* d_ws, size_t ws_size,
                              hipStream_t stream) {
    const float* x      = (const float*)d_in[0];
    const float* gamma  = (const float*)d_in[1];
    const float* beta   = (const float*)d_in[2];
    const float* W_in   = (const float*)d_in[3];
    const float* conv_w = (const float*)d_in[4];
    const float* conv_b = (const float*)d_in[5];
    const float* W_x    = (const float*)d_in[6];
    const float* W_dt   = (const float*)d_in[7];
    const float* b_dt   = (const float*)d_in[8];
    const float* A_log  = (const float*)d_in[9];
    const float* D_skip = (const float*)d_in[10];
    const float* W_out  = (const float*)d_in[11];
    float* out = (float*)d_out;

    char* ws = (char*)d_ws;
    const size_t MB = (size_t)1 << 20;
    half_t* Xn16   = (half_t*)(ws + 0);         //  16 MB (8192x1024 f16)
    half_t* WinT   = (half_t*)(ws + 16 * MB);   //   8 MB (4096x1024 f16, W_in^T)
    float*  xz     = (float*) (ws + 24 * MB);   // 128 MB (8192x4096 f32)
    half_t* Xc16   = (half_t*)(ws + 152 * MB);  //  32 MB (8192x2048 f16)
    half_t* WxT    = (half_t*)(ws + 184 * MB);  // 0.5 MB (128x2048 f16, W_x^T pad)
    float*  dbl    = (float*) (ws + 185 * MB);  //   4 MB (8192x128 f32)
    half_t* Dt16   = (half_t*)(ws + 189 * MB);  //   1 MB (8192x64 f16)
    half_t* WdtT   = (half_t*)(ws + 190 * MB);  // .25 MB (2048x64 f16, W_dt^T)
    float*  dtf    = (float*) (ws + 191 * MB);  //  64 MB (8192x2048 f32)
    half_t* Y16    = (half_t*)(ws + 255 * MB);  //  32 MB (8192x2048 f16)
    half_t* WoutT  = (half_t*)(ws + 287 * MB);  //   4 MB (1024x2048 f16, W_out^T)

    // 1) weight transpose+cast to f16 (recomputed every call: capture-safe)
    {
        int n = DMODEL * 2 * DINNER;  // W_in:  K=1024, N=4096
        transpose_cast_kernel<<<(n + 255) / 256, 256, 0, stream>>>(
            W_in, WinT, DMODEL, 2 * DINNER);
        n = DINNER * DMODEL;          // W_out: K=2048, N=1024
        transpose_cast_kernel<<<(n + 255) / 256, 256, 0, stream>>>(
            W_out, WoutT, DINNER, DMODEL);
        n = DTRANK * DINNER;          // W_dt:  K=64,   N=2048
        transpose_cast_kernel<<<(n + 255) / 256, 256, 0, stream>>>(
            W_dt, WdtT, DTRANK, DINNER);
        n = XPROJ_LD * DINNER;
        transpose_pad_wx_kernel<<<(n + 255) / 256, 256, 0, stream>>>(W_x, WxT);
    }

    // 2) LayerNorm -> f16
    layernorm_kernel<<<ROWS, 256, 0, stream>>>(x, gamma, beta, Xn16);

    // 3) in_proj: (8192x1024) x (1024x4096) -> xz f32
    wmma_gemm_kernel<EPI_NONE><<<dim3((2 * DINNER) / BN, ROWS / BM), 256, 0, stream>>>(
        Xn16, WinT, xz, ROWS, 2 * DINNER, DMODEL, DMODEL, DMODEL, 2 * DINNER,
        nullptr, nullptr);

    // 4) depthwise causal conv + SiLU -> Xc16
    {
        size_t n = (size_t)ROWS * DINNER;
        conv_silu_kernel<<<(unsigned)((n + 255) / 256), 256, 0, stream>>>(
            xz, conv_w, conv_b, Xc16);
    }

    // 5) x_proj: (8192x2048) x (2048x128pad) -> dbl f32
    wmma_gemm_kernel<EPI_NONE><<<dim3(XPROJ_LD / BN, ROWS / BM), 256, 0, stream>>>(
        Xc16, WxT, dbl, ROWS, XPROJ_LD, DINNER, DINNER, DINNER, XPROJ_LD,
        nullptr, nullptr);

    // 6) dt_proj: softplus((8192x64) x (64x2048) + b_dt) -> dtf f32
    {
        int n = ROWS * DTRANK;
        extract_dt_kernel<<<(n + 255) / 256, 256, 0, stream>>>(dbl, Dt16);
    }
    wmma_gemm_kernel<EPI_SOFTPLUS_BIAS><<<dim3(DINNER / BN, ROWS / BM), 256, 0, stream>>>(
        Dt16, WdtT, dtf, ROWS, DINNER, DTRANK, DTRANK, DTRANK, DINNER,
        b_dt, nullptr);

    // 7) selective scan + D-skip + SiLU(z) gating -> Y16
    scan_kernel<<<BATCH * (DINNER / 256), 256, 0, stream>>>(
        dtf, Xc16, dbl, xz, A_log, D_skip, Y16);

    // 8) out_proj + residual: (8192x2048) x (2048x1024) + x -> d_out
    wmma_gemm_kernel<EPI_RESID><<<dim3(DMODEL / BN, ROWS / BM), 256, 0, stream>>>(
        Y16, WoutT, out, ROWS, DMODEL, DINNER, DINNER, DINNER, DMODEL,
        nullptr, x);
}
